// LDMMolDesignRAG_77008763617335
// MI455X (gfx1250) — matrix operand order repaired and checked by reading, not compiled, not run on hardware
//
#include <hip/hip_runtime.h>
#include <hip/hip_bf16.h>
#include <math.h>

#define HID 128
#define NBLOCKS 50000
#define NATOMS (NBLOCKS * 8)
#define NEDGES 800000

typedef __attribute__((ext_vector_type(16))) __bf16 v16bf;
typedef __attribute__((ext_vector_type(8)))  float  v8f;

__device__ __forceinline__ v8f wmma_bf16(v16bf a, v16bf b, v8f c) {
    return __builtin_amdgcn_wmma_f32_16x16x32_bf16(false, a, false, b, (short)0, c, false, false);
}

// B fragment: 32x16 bf16 tile from n-major (transposed) weight matrix.
// Lane n = lane&15 column; K chunks [half*8, half*8+7] and [16+half*8, ...+7].
__device__ __forceinline__ v16bf load_b_glb(const __bf16* __restrict__ wT, int ldk,
                                            int ncol, int kbase, int lane) {
    const __bf16* p = wT + (size_t)(ncol + (lane & 15)) * ldk + kbase + ((lane >> 4) << 3);
    union { uint4 u[2]; v16bf v; } t;
    t.u[0] = *(const uint4*)p;
    t.u[1] = *(const uint4*)(p + 16);
    return t.v;
}

// A fragment: 16x32 bf16 tile from row-major LDS tile (leading dim ld).
// Lane row m = lane&15; K chunks [half*8, half*8+7] and [16+half*8, ...+7].
__device__ __forceinline__ v16bf load_a_lds(const __bf16* t, int ld, int kbase, int lane) {
    int m = lane & 15, half = lane >> 4;
    const __bf16* p = t + (size_t)m * ld + kbase + half * 8;
    union { uint4 u[2]; v16bf v; } r;
    r.u[0] = *(const uint4*)p;
    r.u[1] = *(const uint4*)(p + 16);
    return r.v;
}

// ---------------- prep kernels ----------------
__global__ void zero_f4(float4* p, int n4) {
    int i = blockIdx.x * blockDim.x + threadIdx.x;
    if (i < n4) p[i] = make_float4(0.f, 0.f, 0.f, 0.f);
}

// dst[n*K + k] = bf16(src[k*N + n])   (K x N row-major -> N x K transposed bf16)
__global__ void conv_transpose_bf16(const float* __restrict__ src, __bf16* __restrict__ dst,
                                    int K, int N) {
    int i = blockIdx.x * blockDim.x + threadIdx.x;
    if (i >= K * N) return;
    int k = i / N, n = i % N;
    dst[(size_t)n * K + k] = (__bf16)src[i];
}

// ---------------- edge aggregation ----------------
__global__ void __launch_bounds__(256) edge_agg(
        const int* __restrict__ bonds, const int* __restrict__ A,
        const int* __restrict__ gmask,
        const float* __restrict__ atom_embed, const float* __restrict__ bond_embed,
        float* __restrict__ agg) {
    int t = blockIdx.x * 256 + threadIdx.x;
    int e = t >> 7;      // 128 threads per edge, one channel each
    int c = t & 127;
    if (e >= NEDGES) return;
    int s  = bonds[e * 3 + 0];
    int d  = bonds[e * 3 + 1];
    int bt = bonds[e * 3 + 2];
    if (gmask[s >> 3] || gmask[d >> 3]) return;   // valid = ctx[src] & ctx[dst]
    float v = atom_embed[A[s] * HID + c] + bond_embed[bt * HID + c];
    v = v > 0.f ? v : 0.f;
    unsafeAtomicAdd(&agg[(size_t)d * HID + c], v);
}

// ---------------- GINE MLP + block pooling ----------------
__global__ void __launch_bounds__(256) gine_kernel(
        const int* __restrict__ A, const int* __restrict__ gmask,
        const float* __restrict__ atom_embed, const float* __restrict__ agg,
        const __bf16* __restrict__ w1T, const float* __restrict__ b1,
        const __bf16* __restrict__ w2T, const float* __restrict__ b2,
        float* __restrict__ topo) {
    __shared__ __align__(16) __bf16 yT[8][16 * HID];   // per-wave relu output tile
    int lane = threadIdx.x & 31;
    int wave = threadIdx.x >> 5;
    int rowBase = blockIdx.x * 128 + wave * 16;
    int m = lane & 15, half = lane >> 4;
    int row = rowBase + m;
    const float* emb = atom_embed + (size_t)A[row] * HID;
    const float* g   = agg + (size_t)row * HID;
    __bf16* y = yT[wave];

    // ---- GEMM1: (Hn + agg) @ w1 ----
    v8f acc[8];
#pragma unroll
    for (int nt = 0; nt < 8; nt++)
#pragma unroll
        for (int j = 0; j < 8; j++) acc[nt][j] = 0.f;

#pragma unroll
    for (int ks = 0; ks < 4; ks++) {
        v16bf a;
        int k0 = ks * 32 + half * 8;
#pragma unroll
        for (int ch = 0; ch < 2; ch++) {
            int kk = k0 + ch * 16;
            float4 e0 = *(const float4*)(emb + kk);
            float4 e1 = *(const float4*)(emb + kk + 4);
            float4 g0 = *(const float4*)(g + kk);
            float4 g1 = *(const float4*)(g + kk + 4);
            a[ch * 8 + 0] = (__bf16)(e0.x + g0.x);
            a[ch * 8 + 1] = (__bf16)(e0.y + g0.y);
            a[ch * 8 + 2] = (__bf16)(e0.z + g0.z);
            a[ch * 8 + 3] = (__bf16)(e0.w + g0.w);
            a[ch * 8 + 4] = (__bf16)(e1.x + g1.x);
            a[ch * 8 + 5] = (__bf16)(e1.y + g1.y);
            a[ch * 8 + 6] = (__bf16)(e1.z + g1.z);
            a[ch * 8 + 7] = (__bf16)(e1.w + g1.w);
        }
#pragma unroll
        for (int nt = 0; nt < 8; nt++) {
            v16bf b = load_b_glb(w1T, HID, nt * 16, ks * 32, lane);
            acc[nt] = wmma_bf16(a, b, acc[nt]);
        }
    }
    // bias + relu -> bf16 LDS tile (C layout: vgpr j holds rows j / j+8)
#pragma unroll
    for (int nt = 0; nt < 8; nt++)
#pragma unroll
        for (int j = 0; j < 8; j++) {
            int mm = half ? (j + 8) : j;
            int col = nt * 16 + m;
            float v = acc[nt][j] + b1[col];
            y[mm * HID + col] = (__bf16)(v > 0.f ? v : 0.f);
        }

    // ---- GEMM2: relu_out @ w2 ----
    v8f acc2[8];
#pragma unroll
    for (int nt = 0; nt < 8; nt++)
#pragma unroll
        for (int j = 0; j < 8; j++) acc2[nt][j] = 0.f;
#pragma unroll
    for (int ks = 0; ks < 4; ks++) {
        v16bf a = load_a_lds(y, HID, ks * 32, lane);
#pragma unroll
        for (int nt = 0; nt < 8; nt++) {
            v16bf b = load_b_glb(w2T, HID, nt * 16, ks * 32, lane);
            acc2[nt] = wmma_bf16(a, b, acc2[nt]);
        }
    }

    // ---- block pooling: rows 0-7 -> block blkA (lanes 0-15), rows 8-15 -> blkB (lanes 16-31)
    int blk = (rowBase >> 3) + half;
    bool masked = gmask[blk] != 0;
    const float scale = 0.35355339059327373f;   // 1/sqrt(8)
#pragma unroll
    for (int nt = 0; nt < 8; nt++) {
        float s = 0.f;
#pragma unroll
        for (int j = 0; j < 8; j++) s += acc2[nt][j];
        int col = nt * 16 + m;
        s += 8.f * b2[col];                     // bias added to each of 8 rows
        topo[(size_t)blk * HID + col] = masked ? 0.f : s * scale;
    }
}

// ---------------- block-level conditioning MLP ----------------
__global__ void __launch_bounds__(128) mlp_kernel(
        const int* __restrict__ gmask, const int* __restrict__ pos_ids,
        const int* __restrict__ is_aa, const float* __restrict__ crand,
        const float* __restrict__ aa_embed, const float* __restrict__ topo,
        const __bf16* __restrict__ w0T, const float* __restrict__ b0,
        const __bf16* __restrict__ w1T, const float* __restrict__ b1,
        const __bf16* __restrict__ w2T, const float* __restrict__ b2,
        float* __restrict__ out) {
    __shared__ __align__(16) __bf16 xT[4][16 * 384];   // 48 KB
    __shared__ __align__(16) __bf16 yT[4][16 * HID];   // 16 KB
    int lane = threadIdx.x & 31;
    int wave = threadIdx.x >> 5;
    int rowBase = blockIdx.x * 64 + wave * 16;
    __bf16* x = xT[wave];
    __bf16* y = yT[wave];
    int m = lane & 15, half = lane >> 4;
    const float LOG1E4_64 = 0.14391156855f;    // ln(10000)/64

    // build x = [pos_e | topo | is_aa_e] as bf16
    for (int idx = lane; idx < 16 * 384; idx += 32) {
        int r = idx / 384, col = idx - r * 384;
        int blk = rowBase + r;
        int bc = blk < NBLOCKS ? blk : (NBLOCKS - 1);
        float v;
        if (col < 128) {
            int j = col & 63;
            float ang = (float)pos_ids[bc] * __expf(-LOG1E4_64 * (float)j);
            v = (col < 64) ? __sinf(ang) : __cosf(ang);
        } else if (col < 256) {
            v = topo[(size_t)bc * HID + (col - 128)];   // mask already applied
        } else {
            bool corrupt = (gmask[bc] != 0) && (crand[bc] < 0.1f);
            int ia = corrupt ? 0 : is_aa[bc];
            v = aa_embed[ia * HID + (col - 256)];
        }
        x[idx] = (__bf16)v;
    }

    v8f acc[8];
    // layer 0: K = 384
#pragma unroll
    for (int nt = 0; nt < 8; nt++)
#pragma unroll
        for (int j = 0; j < 8; j++) acc[nt][j] = 0.f;
    for (int ks = 0; ks < 12; ks++) {
        v16bf a = load_a_lds(x, 384, ks * 32, lane);
#pragma unroll
        for (int nt = 0; nt < 8; nt++) {
            v16bf b = load_b_glb(w0T, 384, nt * 16, ks * 32, lane);
            acc[nt] = wmma_bf16(a, b, acc[nt]);
        }
    }
#pragma unroll
    for (int nt = 0; nt < 8; nt++)
#pragma unroll
        for (int j = 0; j < 8; j++) {
            int mm = half ? (j + 8) : j;
            int col = nt * 16 + m;
            float v = acc[nt][j] + b0[col];
            y[mm * HID + col] = (__bf16)(v > 0.f ? v : 0.f);
        }

    // layer 1: K = 128 (write relu result back into x region, ld = 128)
#pragma unroll
    for (int nt = 0; nt < 8; nt++)
#pragma unroll
        for (int j = 0; j < 8; j++) acc[nt][j] = 0.f;
#pragma unroll
    for (int ks = 0; ks < 4; ks++) {
        v16bf a = load_a_lds(y, HID, ks * 32, lane);
#pragma unroll
        for (int nt = 0; nt < 8; nt++) {
            v16bf b = load_b_glb(w1T, HID, nt * 16, ks * 32, lane);
            acc[nt] = wmma_bf16(a, b, acc[nt]);
        }
    }
#pragma unroll
    for (int nt = 0; nt < 8; nt++)
#pragma unroll
        for (int j = 0; j < 8; j++) {
            int mm = half ? (j + 8) : j;
            int col = nt * 16 + m;
            float v = acc[nt][j] + b1[col];
            x[mm * HID + col] = (__bf16)(v > 0.f ? v : 0.f);
        }

    // layer 2: K = 128 -> fp32 output
#pragma unroll
    for (int nt = 0; nt < 8; nt++)
#pragma unroll
        for (int j = 0; j < 8; j++) acc[nt][j] = 0.f;
#pragma unroll
    for (int ks = 0; ks < 4; ks++) {
        v16bf a = load_a_lds(x, HID, ks * 32, lane);
#pragma unroll
        for (int nt = 0; nt < 8; nt++) {
            v16bf b = load_b_glb(w2T, HID, nt * 16, ks * 32, lane);
            acc[nt] = wmma_bf16(a, b, acc[nt]);
        }
    }
#pragma unroll
    for (int nt = 0; nt < 8; nt++)
#pragma unroll
        for (int j = 0; j < 8; j++) {
            int mm = half ? (j + 8) : j;
            int r = rowBase + mm;
            int col = nt * 16 + m;
            if (r < NBLOCKS) out[(size_t)r * HID + col] = acc[nt][j] + b2[col];
        }
}

extern "C" void kernel_launch(void* const* d_in, const int* in_sizes, int n_in,
                              void* d_out, int out_size, void* d_ws, size_t ws_size,
                              hipStream_t stream) {
    const int*   A          = (const int*)d_in[0];
    const int*   bonds      = (const int*)d_in[1];
    // d_in[2] = block_ids (implicit: i/8)
    const int*   gmask      = (const int*)d_in[3];
    const int*   pos_ids    = (const int*)d_in[4];
    const int*   is_aa      = (const int*)d_in[5];
    const float* crand      = (const float*)d_in[6];
    const float* atom_embed = (const float*)d_in[7];
    const float* bond_embed = (const float*)d_in[8];
    const float* aa_embed   = (const float*)d_in[9];
    const float* gine_w1    = (const float*)d_in[10];
    const float* gine_b1    = (const float*)d_in[11];
    const float* gine_w2    = (const float*)d_in[12];
    const float* gine_b2    = (const float*)d_in[13];
    const float* mlp_w0     = (const float*)d_in[14];
    const float* mlp_b0     = (const float*)d_in[15];
    const float* mlp_w1     = (const float*)d_in[16];
    const float* mlp_b1     = (const float*)d_in[17];
    const float* mlp_w2     = (const float*)d_in[18];
    const float* mlp_b2     = (const float*)d_in[19];
    float* out = (float*)d_out;

    char* ws = (char*)d_ws;
    size_t AGG_BYTES  = (size_t)NATOMS * HID * sizeof(float);    // 204.8 MB
    size_t TOPO_BYTES = (size_t)NBLOCKS * HID * sizeof(float);   // 25.6 MB
    float*  agg  = (float*)ws;
    float*  topo = (float*)(ws + AGG_BYTES);
    __bf16* w1T  = (__bf16*)(ws + AGG_BYTES + TOPO_BYTES);
    __bf16* w2T  = w1T + 128 * 128;
    __bf16* w0T  = w2T + 128 * 128;          // 128 x 384
    __bf16* mw1T = w0T + 384 * 128;
    __bf16* mw2T = mw1T + 128 * 128;

    // 1) weight prep: fp32 row-major -> bf16 transposed (n-major, contiguous K)
    conv_transpose_bf16<<<(128 * 128 + 255) / 256, 256, 0, stream>>>(gine_w1, w1T, 128, 128);
    conv_transpose_bf16<<<(128 * 128 + 255) / 256, 256, 0, stream>>>(gine_w2, w2T, 128, 128);
    conv_transpose_bf16<<<(384 * 128 + 255) / 256, 256, 0, stream>>>(mlp_w0,  w0T, 384, 128);
    conv_transpose_bf16<<<(128 * 128 + 255) / 256, 256, 0, stream>>>(mlp_w1,  mw1T, 128, 128);
    conv_transpose_bf16<<<(128 * 128 + 255) / 256, 256, 0, stream>>>(mlp_w2,  mw2T, 128, 128);

    // 2) zero edge-aggregation buffer
    int n4 = NATOMS * HID / 4;
    zero_f4<<<n4 / 256, 256, 0, stream>>>((float4*)agg, n4);

    // 3) edge message scatter (native fp32 atomics)
    edge_agg<<<(NEDGES * 128) / 256, 256, 0, stream>>>(bonds, A, gmask,
                                                       atom_embed, bond_embed, agg);

    // 4) GINE MLP (2x bf16 WMMA GEMM) fused with 8-atom block pooling -> topo
    gine_kernel<<<NATOMS / 128, 256, 0, stream>>>(A, gmask, atom_embed, agg,
                                                  w1T, gine_b1, w2T, gine_b2, topo);

    // 5) conditioning MLP (3x bf16 WMMA GEMM) -> cond
    mlp_kernel<<<(NBLOCKS + 63) / 64, 128, 0, stream>>>(gmask, pos_ids, is_aa, crand,
                                                        aa_embed, topo,
                                                        w0T, mlp_b0, mw1T, mlp_b1,
                                                        mw2T, mlp_b2, out);
}
